// LiDAREncoder_74663711474025
// MI455X (gfx1250) — matrix operand order repaired and checked by reading, not compiled, not run on hardware
//
#include <hip/hip_runtime.h>

// ---------------- problem constants (from reference) ----------------
#define HEADS   8
#define LEVELS  3
#define POINTS  4
#define CDIM    256
#define HDIM    32
#define NQ      5376
#define NVAL    5376
#define BATCH   2
#define NOFF    192           // TP*2
#define NATT    96            // TP
#define NCAT    288           // NOFF + NATT
#define KDIM    256           // inner dim of every GEMM

// Use CDNA5 async global->LDS staging (GLOBAL_LOAD_ASYNC_TO_LDS_B128, ASYNCcnt)
#define USE_ASYNC_LDS 1

typedef __attribute__((ext_vector_type(16))) __bf16         v16bf;
typedef __attribute__((ext_vector_type(8)))  float          v8f;
typedef __attribute__((ext_vector_type(8)))  unsigned short v8us;
typedef __attribute__((ext_vector_type(16))) unsigned short v16us;

__device__ __forceinline__ unsigned short f2bf(float f) {
  unsigned int x = __float_as_uint(f);
  unsigned int r = x + 0x7fffu + ((x >> 16) & 1u);   // round-to-nearest-even
  return (unsigned short)(r >> 16);
}

// ---------------- vectorized f32 -> bf16 (8 elements / thread) ----------------
__global__ void cvt8_f32_bf16(const float* __restrict__ src,
                              unsigned short* __restrict__ dst, int n8) {
  int i = blockIdx.x * blockDim.x + threadIdx.x;
  if (i >= n8) return;
  v8f f = ((const v8f*)src)[i];
  v8us o;
#pragma unroll
  for (int e = 0; e < 8; ++e) o[e] = f2bf(f[e]);
  ((v8us*)dst)[i] = o;
}

// ---------------- GEMM: Out[M,N] = A[M,K]_bf16 * W[N,K]_bf16^T + bias[N] ----------------
// Block = 256 thr = 8 waves stacked along M -> block tile 128(M) x 32(N).
// Wave tile = 16(M) x 32(N): two 16x16 f32 accumulators, A fragment reused 2x.
// Weight tile (32 rows x 256 K bf16 = 16 KB) staged once into LDS via
// global_load_async_to_lds_b128 (ASYNCcnt) then read via ds_load.
// Fragment layouts used here, K contiguous and same order in A and B (l16 = lane&15, hi = lane>>4):
//   A: row l16, K = hi*16 .. hi*16+15 (contiguous 32B)
//   B: W row (16*j + l16), K = hi*16 .. hi*16+15 (contiguous 32B)
//   C/D elem r: (M = row0 + 8*hi + r, N = col + l16)
__global__ __launch_bounds__(256)
void gemm_bf16_lds(const unsigned short* __restrict__ A,
                   const unsigned short* __restrict__ W,
                   const float* __restrict__ bias,
                   float* __restrict__ Out,
                   int N) {
  __shared__ unsigned short Wt[32 * KDIM];   // 16 KB

  const int t    = threadIdx.x;
  const int lane = t & 31;
  const int wave = t >> 5;
  const int l16  = lane & 15;
  const int hi   = lane >> 4;

  const int row0 = blockIdx.x * 128 + wave * 16;
  const int colb = blockIdx.y * 32;

  // ---- stage 32x256 bf16 weight tile into LDS ----
  {
    const unsigned short* gsrc = W + (size_t)colb * KDIM;
#if USE_ASYNC_LDS
    typedef __attribute__((address_space(3))) unsigned short lds_us_t;
    unsigned ldsbase = (unsigned)(size_t)(lds_us_t*)Wt;
#pragma unroll
    for (int i = 0; i < 4; ++i) {
      const int idx = t + i * 256;                 // 1024 x 16B chunks
      unsigned lofs = ldsbase + idx * 16;
      const unsigned short* g = gsrc + idx * 8;
      asm volatile("global_load_async_to_lds_b128 %0, %1, off"
                   :: "v"(lofs), "v"(g) : "memory");
    }
    asm volatile("s_wait_asynccnt 0" ::: "memory");
#else
#pragma unroll
    for (int i = 0; i < 4; ++i) {
      const int idx = t + i * 256;
      ((v8us*)Wt)[idx] = ((const v8us*)gsrc)[idx];
    }
#endif
  }
  __syncthreads();

  const unsigned short* ap  = A  + (size_t)(row0 + l16) * KDIM + hi * 16;
  const unsigned short* wt0 = Wt + (size_t)(l16)      * KDIM + hi * 16;  // rows 0..15
  const unsigned short* wt1 = Wt + (size_t)(16 + l16) * KDIM + hi * 16;  // rows 16..31

  v8f acc0 = {};
  v8f acc1 = {};
#pragma unroll
  for (int k0 = 0; k0 < KDIM; k0 += 32) {
    v16us au;
    *(v8us*)&au         = *(const v8us*)(ap + k0);        // K = hi*16 + 0..7
    *(((v8us*)&au) + 1) = *(const v8us*)(ap + k0 + 8);    // K = hi*16 + 8..15
    v16bf av = __builtin_bit_cast(v16bf, au);

    v16us w0 = *(const v16us*)(wt0 + k0);                 // ds_load 32B
    v16us w1 = *(const v16us*)(wt1 + k0);

    acc0 = __builtin_amdgcn_wmma_f32_16x16x32_bf16(false, av, false,
             __builtin_bit_cast(v16bf, w0), (short)0, acc0, false, false);
    acc1 = __builtin_amdgcn_wmma_f32_16x16x32_bf16(false, av, false,
             __builtin_bit_cast(v16bf, w1), (short)0, acc1, false, false);
  }

  const float b0 = bias[colb + l16];
  const float b1 = bias[colb + 16 + l16];
#pragma unroll
  for (int r = 0; r < 8; ++r) {
    const int m = row0 + 8 * hi + r;
    Out[(size_t)m * N + colb + l16]      = acc0[r] + b0;
    Out[(size_t)m * N + colb + 16 + l16] = acc1[r] + b1;
  }
}

// ---------------- deformable sampling: one wave per (b, q, head); lane = channel ----------------
__global__ __launch_bounds__(256)
void msda_sample(const float* __restrict__ vproj,     // (B*NV, 256), c = h*32+hd
                 const float* __restrict__ oa,        // (B*NQ, 288): [0..191]=off, [192..287]=attn
                 const float* __restrict__ refp,      // (B*NQ, 2)
                 unsigned short* __restrict__ houtb)  // (B*NQ, 256) bf16
{
  const int gw   = (blockIdx.x * blockDim.x + threadIdx.x) >> 5;
  const int lane = threadIdx.x & 31;
  const int h  = gw & (HEADS - 1);
  const int bq = gw >> 3;                 // b*NQ + q
  const int b  = bq / NQ;

  const float* oaq = oa + (size_t)bq * NCAT;
  const float rx = refp[bq * 2 + 0];
  const float ry = refp[bq * 2 + 1];

  // softmax over the 12 (level,point) logits of this head
  float aw[LEVELS * POINTS];
  float mx = -1e30f;
#pragma unroll
  for (int j = 0; j < LEVELS * POINTS; ++j) {
    aw[j] = oaq[NOFF + h * (LEVELS * POINTS) + j];
    mx = fmaxf(mx, aw[j]);
  }
  float s = 0.f;
#pragma unroll
  for (int j = 0; j < LEVELS * POINTS; ++j) { aw[j] = __expf(aw[j] - mx); s += aw[j]; }
  const float inv = 1.f / s;

  const float* vb = vproj + (size_t)b * NVAL * CDIM + h * HDIM + lane;

  float acc = 0.f;
  int start = 0;
  const int HWs[LEVELS][2] = {{64, 64}, {32, 32}, {16, 16}};
#pragma unroll
  for (int l = 0; l < LEVELS; ++l) {
    const int Hh = HWs[l][0], Ww = HWs[l][1];
#pragma unroll
    for (int p = 0; p < POINTS; ++p) {
      const float ox = oaq[h * (LEVELS * POINTS * 2) + l * (POINTS * 2) + p * 2 + 0];
      const float oy = oaq[h * (LEVELS * POINTS * 2) + l * (POINTS * 2) + p * 2 + 1];
      const float lx = fminf(fmaxf(rx + ox, 0.f), 1.f);
      const float ly = fminf(fmaxf(ry + oy, 0.f), 1.f);
      const float x = lx * (float)Ww - 0.5f;
      const float y = ly * (float)Hh - 0.5f;
      const float x0f = floorf(x), y0f = floorf(y);
      const float wx1 = x - x0f, wy1 = y - y0f;
      const int x0 = (int)x0f, y0 = (int)y0f;
      const float ap = aw[l * POINTS + p] * inv;
#pragma unroll
      for (int dy = 0; dy < 2; ++dy) {
#pragma unroll
        for (int dx = 0; dx < 2; ++dx) {
          const int ix = x0 + dx, iy = y0 + dy;
          if (ix < 0 || ix >= Ww || iy < 0 || iy >= Hh) continue;
          const float wgt = (dx ? wx1 : 1.f - wx1) * (dy ? wy1 : 1.f - wy1);
          acc += ap * wgt * vb[(size_t)(start + iy * Ww + ix) * CDIM];
        }
      }
    }
    start += Hh * Ww;
  }
  houtb[(size_t)bq * CDIM + h * HDIM + lane] = f2bf(acc);
}

// ---------------- launch ----------------
extern "C" void kernel_launch(void* const* d_in, const int* in_sizes, int n_in,
                              void* d_out, int out_size, void* d_ws, size_t ws_size,
                              hipStream_t stream) {
  (void)in_sizes; (void)n_in; (void)out_size; (void)ws_size;

  const float* query = (const float*)d_in[0];
  const float* refp  = (const float*)d_in[1];
  const float* value = (const float*)d_in[2];
  const float* Wv    = (const float*)d_in[3];
  const float* bv    = (const float*)d_in[4];
  const float* Woff  = (const float*)d_in[5];
  const float* boff  = (const float*)d_in[6];
  const float* Wa    = (const float*)d_in[7];
  const float* ba    = (const float*)d_in[8];
  const float* Wo    = (const float*)d_in[9];
  const float* bo    = (const float*)d_in[10];

  char*  ws  = (char*)d_ws;
  size_t off = 0;
  auto alloc = [&](size_t bytes) -> void* {
    void* p = ws + off;
    off += (bytes + 255) & ~(size_t)255;
    return p;
  };

  const size_t ACT = (size_t)BATCH * NQ * CDIM;       // 2752512

  unsigned short* qb    = (unsigned short*)alloc(ACT * 2);
  unsigned short* valb  = (unsigned short*)alloc(ACT * 2);
  unsigned short* houtb = (unsigned short*)alloc(ACT * 2);
  unsigned short* Wvb   = (unsigned short*)alloc((size_t)CDIM * CDIM * 2);
  unsigned short* Wcatb = (unsigned short*)alloc((size_t)NCAT * CDIM * 2);
  unsigned short* Wob   = (unsigned short*)alloc((size_t)CDIM * CDIM * 2);
  float* bcat  = (float*)alloc((size_t)NCAT * 4);
  float* vproj = (float*)alloc(ACT * 4);
  float* oa    = (float*)alloc((size_t)BATCH * NQ * NCAT * 4);

  dim3 blk(256, 1, 1);
  auto cvtgrid = [](size_t n8) { return (unsigned)((n8 + 255) / 256); };

  // bf16 conversions (activations + weights); Wcat = [Woff ; Wa], bcat = [boff ; ba]
  cvt8_f32_bf16<<<cvtgrid(ACT / 8), blk, 0, stream>>>(query, qb,   (int)(ACT / 8));
  cvt8_f32_bf16<<<cvtgrid(ACT / 8), blk, 0, stream>>>(value, valb, (int)(ACT / 8));
  cvt8_f32_bf16<<<cvtgrid(CDIM * CDIM / 8), blk, 0, stream>>>(Wv, Wvb, CDIM * CDIM / 8);
  cvt8_f32_bf16<<<cvtgrid(NOFF * CDIM / 8), blk, 0, stream>>>(Woff, Wcatb, NOFF * CDIM / 8);
  cvt8_f32_bf16<<<cvtgrid(NATT * CDIM / 8), blk, 0, stream>>>(Wa, Wcatb + NOFF * CDIM, NATT * CDIM / 8);
  cvt8_f32_bf16<<<cvtgrid(CDIM * CDIM / 8), blk, 0, stream>>>(Wo, Wob, CDIM * CDIM / 8);
  hipMemcpyAsync(bcat,        boff, NOFF * 4, hipMemcpyDeviceToDevice, stream);
  hipMemcpyAsync(bcat + NOFF, ba,   NATT * 4, hipMemcpyDeviceToDevice, stream);

  const int Mrows = BATCH * NQ;          // 10752, divisible by 128

  // 1) value projection: vproj = value @ Wv^T + bv
  dim3 g1(Mrows / 128, CDIM / 32, 1);
  gemm_bf16_lds<<<g1, blk, 0, stream>>>(valb, Wvb, bv, vproj, CDIM);

  // 2) offsets + attention logits: oa = query @ [Woff;Wa]^T + [boff;ba]
  dim3 g2(Mrows / 128, NCAT / 32, 1);
  gemm_bf16_lds<<<g2, blk, 0, stream>>>(qb, Wcatb, bcat, oa, NCAT);

  // 3) softmax + bilinear deformable sampling (one wave per (b,q,head)); emits bf16
  const int waves = BATCH * NQ * HEADS;  // 86016
  msda_sample<<<waves * 32 / 256, blk, 0, stream>>>(vproj, oa, refp, houtb);

  // 4) output projection: out = hout @ Wo^T + bo
  dim3 g3(Mrows / 128, CDIM / 32, 1);
  gemm_bf16_lds<<<g3, blk, 0, stream>>>(houtb, Wob, bo, (float*)d_out, CDIM);
}
